// MultiHeaderModel_53644141527060
// MI455X (gfx1250) — matrix operand order repaired and checked by reading, not compiled, not run on hardware
//
#include <hip/hip_runtime.h>
#include <hip/hip_bf16.h>

// ---------------------------------------------------------------------------
// MultiHeaderModel for MI455X (gfx1250, wave32, WMMA).
//
// Pipeline (all on `stream`, sequential deps):
//   K0 prep   : transpose/convert Wx_{f,b} -> f16 [80][256]; fold Wuv@Wr into
//               W1,W2 (f16, [n][k]); bias2 = buv@Wr + br.
//   K1 embed  : gather char/word tables -> f16 embed [4096,256]
//   K2 gates  : WMMA f16 GEMM [4096,256]x[256,80] + bias -> gx (f32)
//   K3 lstm   : 32 waves, one per (batch,dir); 256-step recurrence in LDS
//   K4 sent   : sent=[h_f|h_b]; entity_logits = sent@We+be -> d_out[0:122880)
//   K5 susv   : su/sv = relu(rel_enc@Wu/Wv + b) -> f16
//   K6 PQ     : WMMA f16 GEMMs P=su@W1, Q=sv@W2 (f32)
//   K7 final  : rel_logits[b,i,j,:] = softmax(P[b,j,:]+Q[b,i,:]+bias2)
//               (store-bandwidth bound: 256 MB of output)
// ---------------------------------------------------------------------------

typedef __attribute__((ext_vector_type(16))) _Float16 v16h;
typedef __attribute__((ext_vector_type(8)))  float    v8f;

#define Bn 16
#define Ln 256
#define BL 4096          // B*L
#define Dn 256           // 2*EMB
#define Gn 40            // 4*H
#define Hn 10
#define ENT_NUM 30
#define ENT_EMB 64
#define ENC 84
#define RELn 64
#define OUT_ENT (BL * ENT_NUM)   // 122880

// workspace byte offsets (all 256B aligned)
#define OFF_EMBED 0u
#define OFF_WXT   2097152u               // + 4096*256*2
#define OFF_W1T   2138112u               // + 80*256*2
#define OFF_W2T   2146304u               // + 64*64*2
#define OFF_BIAS2 2154496u               // + 64*64*2
#define OFF_GX    2154752u               // + 64*4 (padded)
#define OFF_HF    3465472u               // + 4096*80*4
#define OFF_HB    3629312u               // + 4096*10*4
#define OFF_SENT  3793152u               // + 4096*10*4
#define OFF_SUH   4120832u               // + 4096*20*4
#define OFF_SVH   4645120u               // + 4096*64*2
#define OFF_P     5169408u               // + 4096*64*2
#define OFF_Q     6217984u               // + 4096*64*4

__device__ __forceinline__ float sigmoidf_(float x) {
    return 1.0f / (1.0f + __expf(-x));
}

// ---------------- K0: weight prep ----------------
__global__ void prep_kernel(const float* __restrict__ Wx_f, const float* __restrict__ Wx_b,
                            const float* __restrict__ Wuv, const float* __restrict__ Wr,
                            const float* __restrict__ buv, const float* __restrict__ br,
                            _Float16* __restrict__ WxT, _Float16* __restrict__ W1T,
                            _Float16* __restrict__ W2T, float* __restrict__ bias2) {
    int gid = blockIdx.x * blockDim.x + threadIdx.x;
    if (gid < 80 * 256) {
        int col = gid >> 8, k = gid & 255;
        float v = (col < Gn) ? Wx_f[k * Gn + col] : Wx_b[k * Gn + (col - Gn)];
        WxT[col * 256 + k] = (_Float16)v;           // [N=80][K=256], f16
    } else if (gid < 80 * 256 + 2 * 64 * 64) {
        int t = gid - 80 * 256;
        int which = t >> 12;                         // 0 -> W1 (su side), 1 -> W2 (sv side)
        int idx = t & 4095;
        int n = idx >> 6, k = idx & 63;
        const float* Wsrc = Wuv + (which ? 64 * 64 : 0);
        float acc = 0.f;
        for (int r = 0; r < 64; ++r) acc += Wsrc[k * 64 + r] * Wr[r * 64 + n];
        (which ? W2T : W1T)[n * 64 + k] = (_Float16)acc;   // [N=64][K=64], f16
    } else if (gid < 80 * 256 + 2 * 64 * 64 + 64) {
        int n = gid - (80 * 256 + 2 * 64 * 64);
        float acc = br[n];
        for (int k = 0; k < 64; ++k) acc += buv[k] * Wr[k * 64 + n];
        bias2[n] = acc;
    }
}

// ---------------- K1: embedding gather -> f16 ----------------
__global__ void embed_kernel(const int* __restrict__ char_ids, const int* __restrict__ word_ids,
                             const float* __restrict__ char_table, const float* __restrict__ word_table,
                             _Float16* __restrict__ embed_h) {
    int row = blockIdx.x;            // 0..4095
    int c = threadIdx.x;             // 0..255
    float v = (c < 128) ? char_table[char_ids[row] * 128 + c]
                        : word_table[word_ids[row] * 128 + (c - 128)];
    embed_h[row * 256 + c] = (_Float16)v;
}

// ---------------- K2: gates GEMM via WMMA (f16 in, f32 acc) ----------------
// A: embed_h [4096,256] row-major f16.  B: WxT [80][256] (pre-transposed).
// One wave per 16-row M tile; 5 N tiles (80 cols); K loop 256/32.
__global__ void __launch_bounds__(32) gates_wmma_kernel(
        const _Float16* __restrict__ embed_h, const _Float16* __restrict__ WxT,
        const float* __restrict__ b_f, const float* __restrict__ b_b,
        float* __restrict__ gx) {
    int lane = threadIdx.x;
    int m0 = blockIdx.x * 16;
    int mrow = m0 + (lane & 15);
    int kgrpA = (lane >> 4) * 8;     // A layout: lanes 0-15 K{0..7,16..23}, 16-31 K{8..15,24..31}
    int nlane = lane & 15;
    int khalfB = (lane >> 4) * 8;    // B layout: lanes 0-15 K{0..7,16..23}, 16-31 K{8..15,24..31}

    v8f acc[5] = {};
    for (int kk = 0; kk < 256; kk += 32) {
        union { v16h v; unsigned u[8]; } A;
#pragma unroll
        for (int r = 0; r < 8; ++r) {
            int kA = ((r < 4) ? (2 * r) : (16 + 2 * (r - 4))) + kgrpA;
            A.u[r] = *(const unsigned*)(embed_h + mrow * 256 + kk + kA);
        }
#pragma unroll
        for (int nt = 0; nt < 5; ++nt) {
            union { v16h v; unsigned u[8]; } Bm;
            int n = nt * 16 + nlane;
#pragma unroll
            for (int r = 0; r < 8; ++r)
                Bm.u[r] = *(const unsigned*)(WxT + n * 256 + kk + khalfB + ((r < 4) ? (2 * r) : (16 + 2 * (r - 4))));
            acc[nt] = __builtin_amdgcn_wmma_f32_16x16x32_f16(
                false, A.v, false, Bm.v, (short)0, acc[nt], false, false);
        }
    }
#pragma unroll
    for (int nt = 0; nt < 5; ++nt) {
        int col = nt * 16 + nlane;
        float bias = (col < Gn) ? b_f[col] : b_b[col - Gn];
#pragma unroll
        for (int v = 0; v < 8; ++v) {
            int row = m0 + v + 8 * (lane >> 4);
            gx[row * 80 + col] = acc[nt][v] + bias;
        }
    }
}

// ---------------- K3: masked LSTM recurrence, one wave per (batch, dir) ----------------
__global__ void __launch_bounds__(32) lstm_kernel(
        const float* __restrict__ gx, const float* __restrict__ Wh_f,
        const float* __restrict__ Wh_b, const int* __restrict__ char_ids,
        float* __restrict__ h_out_f, float* __restrict__ h_out_b) {
    int b = blockIdx.x & 15;
    int dir = blockIdx.x >> 4;
    int lane = threadIdx.x;
    const float* Wh = dir ? Wh_b : Wh_f;
    float* hout = dir ? h_out_b : h_out_f;

    __shared__ float hs[Hn], cs[Hn], zs[Gn];
    if (lane < Hn) { hs[lane] = 0.f; cs[lane] = 0.f; }
    __syncthreads();

    for (int step = 0; step < Ln; ++step) {
        int t = dir ? (Ln - 1 - step) : step;
        int row = b * Ln + t;
        for (int u = lane; u < Gn; u += 32) {
            float z = gx[row * 80 + dir * Gn + u];
#pragma unroll
            for (int k = 0; k < Hn; ++k) z += hs[k] * Wh[k * Gn + u];
            zs[u] = z;
        }
        __syncthreads();
        if (lane < Hn) {
            float i = sigmoidf_(zs[lane]);
            float f = sigmoidf_(zs[Hn + lane]);
            float g = tanhf(zs[2 * Hn + lane]);
            float o = sigmoidf_(zs[3 * Hn + lane]);
            float cn = f * cs[lane] + i * g;
            float hn = o * tanhf(cn);
            if (char_ids[row] != 0) { hs[lane] = hn; cs[lane] = cn; }
            hout[row * Hn + lane] = hs[lane];
        }
        __syncthreads();
    }
}

// ---------------- K4: sent concat + entity logits ----------------
__global__ void sent_entity_kernel(const float* __restrict__ h_f, const float* __restrict__ h_b,
                                   const float* __restrict__ We, const float* __restrict__ be,
                                   float* __restrict__ sent, float* __restrict__ out_ent) {
    int gid = blockIdx.x * blockDim.x + threadIdx.x;    // exactly 4096*30
    int row = gid / ENT_NUM;
    int e = gid - row * ENT_NUM;
    float acc = be[e];
#pragma unroll
    for (int k = 0; k < 2 * Hn; ++k) {
        float s = (k < Hn) ? h_f[row * Hn + k] : h_b[row * Hn + (k - Hn)];
        acc += s * We[k * ENT_NUM + e];
    }
    out_ent[row * ENT_NUM + e] = acc;
    if (e < 2 * Hn)
        sent[row * 2 * Hn + e] = (e < Hn) ? h_f[row * Hn + e] : h_b[row * Hn + (e - Hn)];
}

// ---------------- K5: su / sv = relu(rel_enc @ Wu/Wv + b) -> f16 ----------------
__global__ void susv_kernel(const float* __restrict__ sent, const int* __restrict__ entity_ids,
                            const float* __restrict__ ent_table,
                            const float* __restrict__ Wu, const float* __restrict__ bu,
                            const float* __restrict__ Wv, const float* __restrict__ bv,
                            _Float16* __restrict__ su_h, _Float16* __restrict__ sv_h) {
    int gid = blockIdx.x * blockDim.x + threadIdx.x;    // exactly 4096*128
    int row = gid >> 7;
    int c = gid & 127;
    int which = c >> 6;         // 0 -> su, 1 -> sv
    int n = c & 63;
    const float* W = which ? Wv : Wu;
    float acc = which ? bv[n] : bu[n];
#pragma unroll 4
    for (int k = 0; k < 2 * Hn; ++k) acc += sent[row * 2 * Hn + k] * W[k * RELn + n];
    const float* ev = ent_table + entity_ids[row] * ENT_EMB;
#pragma unroll 4
    for (int k = 0; k < ENT_EMB; ++k) acc += ev[k] * W[(2 * Hn + k) * RELn + n];
    acc = fmaxf(acc, 0.f);
    (which ? sv_h : su_h)[row * RELn + n] = (_Float16)acc;
}

// ---------------- K6: P = su@W1, Q = sv@W2 via WMMA ----------------
__global__ void __launch_bounds__(32) pq_wmma_kernel(
        const _Float16* __restrict__ su_h, const _Float16* __restrict__ sv_h,
        const _Float16* __restrict__ W1T, const _Float16* __restrict__ W2T,
        float* __restrict__ P, float* __restrict__ Q) {
    int lane = threadIdx.x;
    int m0 = blockIdx.x * 16;
    int mrow = m0 + (lane & 15);
    int kgrpA = (lane >> 4) * 8;
    int nlane = lane & 15;
    int khalfB = (lane >> 4) * 8;

#pragma unroll
    for (int which = 0; which < 2; ++which) {
        const _Float16* Ain = which ? sv_h : su_h;
        const _Float16* WT = which ? W2T : W1T;
        float* Out = which ? Q : P;
        v8f acc[4] = {};
#pragma unroll
        for (int kk = 0; kk < 64; kk += 32) {
            union { v16h v; unsigned u[8]; } A;
#pragma unroll
            for (int r = 0; r < 8; ++r) {
                int kA = ((r < 4) ? (2 * r) : (16 + 2 * (r - 4))) + kgrpA;
                A.u[r] = *(const unsigned*)(Ain + mrow * 64 + kk + kA);
            }
#pragma unroll
            for (int nt = 0; nt < 4; ++nt) {
                union { v16h v; unsigned u[8]; } Bm;
                int n = nt * 16 + nlane;
#pragma unroll
                for (int r = 0; r < 8; ++r)
                    Bm.u[r] = *(const unsigned*)(WT + n * 64 + kk + khalfB + ((r < 4) ? (2 * r) : (16 + 2 * (r - 4))));
                acc[nt] = __builtin_amdgcn_wmma_f32_16x16x32_f16(
                    false, A.v, false, Bm.v, (short)0, acc[nt], false, false);
            }
        }
#pragma unroll
        for (int nt = 0; nt < 4; ++nt) {
            int col = nt * 16 + nlane;
#pragma unroll
            for (int v = 0; v < 8; ++v) {
                int row = m0 + v + 8 * (lane >> 4);
                Out[row * 64 + col] = acc[nt][v];
            }
        }
    }
}

// ---------------- K7: rel_logits = softmax(P_j + Q_i + bias2) -> 256 MB out ----------------
__global__ void __launch_bounds__(256) final_softmax_kernel(
        const float* __restrict__ P, const float* __restrict__ Q,
        const float* __restrict__ bias2, float* __restrict__ out_rel) {
    int bi = blockIdx.x;     // b*L + i
    int j = threadIdx.x;     // 0..255
    const float4* p4 = (const float4*)(P + (size_t)((bi & ~255) + j) * 64);  // (b*L + j)*64
    const float4* q4 = (const float4*)(Q + (size_t)bi * 64);                 // block-uniform
    const float4* b4 = (const float4*)bias2;

    float l[64];
    float m = -3.402823466e38f;
#pragma unroll
    for (int t = 0; t < 16; ++t) {
        float4 pv = p4[t], qv = q4[t], bv = b4[t];
        float a0 = pv.x + qv.x + bv.x;
        float a1 = pv.y + qv.y + bv.y;
        float a2 = pv.z + qv.z + bv.z;
        float a3 = pv.w + qv.w + bv.w;
        l[4 * t + 0] = a0; l[4 * t + 1] = a1; l[4 * t + 2] = a2; l[4 * t + 3] = a3;
        m = fmaxf(m, fmaxf(fmaxf(a0, a1), fmaxf(a2, a3)));
    }
    float s = 0.f;
#pragma unroll
    for (int k = 0; k < 64; ++k) {
        float e = __expf(l[k] - m);
        l[k] = e;
        s += e;
    }
    float inv = 1.0f / s;
    float4* o4 = (float4*)(out_rel + ((size_t)bi * 256 + j) * 64);
#pragma unroll
    for (int t = 0; t < 16; ++t) {
        float4 o;
        o.x = l[4 * t + 0] * inv;
        o.y = l[4 * t + 1] * inv;
        o.z = l[4 * t + 2] * inv;
        o.w = l[4 * t + 3] * inv;
        o4[t] = o;
    }
}

// ---------------------------------------------------------------------------
extern "C" void kernel_launch(void* const* d_in, const int* in_sizes, int n_in,
                              void* d_out, int out_size, void* d_ws, size_t ws_size,
                              hipStream_t stream) {
    (void)in_sizes; (void)n_in; (void)out_size; (void)ws_size;
    const int*   char_ids   = (const int*)d_in[0];
    const int*   word_ids   = (const int*)d_in[1];
    const int*   entity_ids = (const int*)d_in[2];
    const float* char_table = (const float*)d_in[3];
    const float* word_table = (const float*)d_in[4];
    const float* ent_table  = (const float*)d_in[5];
    const float* Wx_f = (const float*)d_in[6];
    const float* Wh_f = (const float*)d_in[7];
    const float* b_f  = (const float*)d_in[8];
    const float* Wx_b = (const float*)d_in[9];
    const float* Wh_b = (const float*)d_in[10];
    const float* b_b  = (const float*)d_in[11];
    const float* We   = (const float*)d_in[12];
    const float* be   = (const float*)d_in[13];
    const float* Wu   = (const float*)d_in[14];
    const float* bu   = (const float*)d_in[15];
    const float* Wv   = (const float*)d_in[16];
    const float* bv   = (const float*)d_in[17];
    const float* Wuv  = (const float*)d_in[18];
    const float* buv  = (const float*)d_in[19];
    const float* Wr   = (const float*)d_in[20];
    const float* br   = (const float*)d_in[21];

    char* ws = (char*)d_ws;
    _Float16* embed_h = (_Float16*)(ws + OFF_EMBED);
    _Float16* WxT     = (_Float16*)(ws + OFF_WXT);
    _Float16* W1T     = (_Float16*)(ws + OFF_W1T);
    _Float16* W2T     = (_Float16*)(ws + OFF_W2T);
    float*    bias2   = (float*)(ws + OFF_BIAS2);
    float*    gx      = (float*)(ws + OFF_GX);
    float*    h_f     = (float*)(ws + OFF_HF);
    float*    h_b     = (float*)(ws + OFF_HB);
    float*    sent    = (float*)(ws + OFF_SENT);
    _Float16* su_h    = (_Float16*)(ws + OFF_SUH);
    _Float16* sv_h    = (_Float16*)(ws + OFF_SVH);
    float*    Pm      = (float*)(ws + OFF_P);
    float*    Qm      = (float*)(ws + OFF_Q);

    float* out_ent = (float*)d_out;
    float* out_rel = (float*)d_out + OUT_ENT;

    prep_kernel<<<113, 256, 0, stream>>>(Wx_f, Wx_b, Wuv, Wr, buv, br, WxT, W1T, W2T, bias2);
    embed_kernel<<<BL, 256, 0, stream>>>(char_ids, word_ids, char_table, word_table, embed_h);
    gates_wmma_kernel<<<BL / 16, 32, 0, stream>>>(embed_h, WxT, b_f, b_b, gx);
    lstm_kernel<<<32, 32, 0, stream>>>(gx, Wh_f, Wh_b, char_ids, h_f, h_b);
    sent_entity_kernel<<<(BL * ENT_NUM) / 256, 256, 0, stream>>>(h_f, h_b, We, be, sent, out_ent);
    susv_kernel<<<(BL * 128) / 256, 256, 0, stream>>>(sent, entity_ids, ent_table,
                                                      Wu, bu, Wv, bv, su_h, sv_h);
    pq_wmma_kernel<<<BL / 16, 32, 0, stream>>>(su_h, sv_h, W1T, W2T, Pm, Qm);
    final_softmax_kernel<<<BL, 256, 0, stream>>>(Pm, Qm, bias2, out_rel);
}
